// SDSEF_FM_73607149519212
// MI455X (gfx1250) — matrix operand order repaired and checked by reading, not compiled, not run on hardware
//
#include <hip/hip_runtime.h>
#include <math.h>

#define DIM 16
#define HID 128
#define BATCH 1024
#define STEPS 10
#define SAMP_PER_WG 8
#define WAVES 16
#define THREADS (WAVES * 32)
#define NBLK (BATCH / SAMP_PER_WG)

typedef __attribute__((ext_vector_type(16))) __bf16 v16bf;
typedef __attribute__((ext_vector_type(8)))  float  v8f;
typedef __attribute__((ext_vector_type(4)))  unsigned int v4u;

union Vb {
    v16bf v;
    v4u   u4[2];
    unsigned u32[8];
    unsigned short h[16];
};

struct Smem {
    unsigned short wt0[HID * 32];          // layer0 W^T, K padded to 32 (t-row + bias folded)
    unsigned short wt123[3][HID * HID];    // layers 1..3 W^T (bf16)
    unsigned short wt4[DIM * HID];         // layer4 W^T
    unsigned short hbuf[16 * HID];         // primal activations (bf16, row-major [M][K])
    unsigned short thbuf[8 * 16 * HID];    // per-sample tangent activations
    float bias123[3 * HID];
    float bias4[DIM];
    float dbuf[16 * HID];                  // 1 - h^2, all 16 primal rows (no lane divergence)
    float z[256], zs[256], ka[256];        // primal RK4 state (16 rows x 16)
    float Tz[2048], Tzs[2048], Tka[2048];  // tangent RK4 state (8 x 16 x 16)
    float vout[256];
    float Tvout[2048];
};

__device__ __forceinline__ unsigned short f2bf(float f) {
    unsigned u = __float_as_uint(f);
    u += 0x7FFFu + ((u >> 16) & 1u);
    return (unsigned short)(u >> 16);
}

// Pack two f32 -> packed bf16x2.
// Preferred: single v_cvt_pk_bf16_f32 (if builtin exists).
// Fallback: +0x8000 round bias per element, then one v_perm_b32 to merge the
// two high halves ({hi[31:16], lo[31:16]}): 3 VALU per pair.
__device__ __forceinline__ unsigned pack2bf(float lo, float hi) {
#if defined(__has_builtin)
#if __has_builtin(__builtin_amdgcn_cvt_pk_bf16_f32)
    typedef __attribute__((ext_vector_type(2))) __bf16 v2bf;
    v2bf p = __builtin_amdgcn_cvt_pk_bf16_f32(lo, hi);
    return __builtin_bit_cast(unsigned, p);
#else
    unsigned ul = __float_as_uint(lo) + 0x8000u;
    unsigned uh = __float_as_uint(hi) + 0x8000u;
    return __builtin_amdgcn_perm(uh, ul, 0x07060302u);
#endif
#else
    unsigned ul = __float_as_uint(lo) + 0x8000u;
    unsigned uh = __float_as_uint(hi) + 0x8000u;
    return __builtin_amdgcn_perm(uh, ul, 0x07060302u);
#endif
}

// Branchless tanh: native V_TANH_F32 on gfx1250.
__device__ __forceinline__ float fast_tanh(float x) {
#if defined(__has_builtin)
#if __has_builtin(__builtin_amdgcn_tanhf)
    return __builtin_amdgcn_tanhf(x);
#else
    float e = __builtin_amdgcn_exp2f(x * 2.8853900817779268f); // 2*log2(e)
    return 1.0f - 2.0f * __builtin_amdgcn_rcpf(1.0f + e);
#endif
#else
    float e = __builtin_amdgcn_exp2f(x * 2.8853900817779268f);
    return 1.0f - 2.0f * __builtin_amdgcn_rcpf(1.0f + e);
#endif
}

__device__ __forceinline__ v8f splat8(float v) {
    v8f r;
#pragma unroll
    for (int i = 0; i < 8; ++i) r[i] = v;
    return r;
}

// A operand (16xK row-major bf16 in LDS) -> v16bf per ISA 16-bit A layout:
// lanes 0-15 hold K = kb*32 + {0..7, 16..23}; lanes 16-31 hold {8..15, 24..31}.
__device__ __forceinline__ v16bf loadA(const unsigned short* base, int pitch, int kb, int lane) {
    const unsigned short* p = base + (lane & 15) * pitch + kb * 32 + ((lane & 16) ? 8 : 0);
    Vb v;
    v.u4[0] = *(const v4u*)(p);
    v.u4[1] = *(const v4u*)(p + 16);
    return v.v;
}

// B operand (W^T stored [N][K] row-major bf16): lane L holds N = nt*16 + L%16,
// K = kb*32 + (L<16 ? {0..7, 16..23} : {8..15, 24..31}) (same order as A).
__device__ __forceinline__ v16bf loadB(const unsigned short* base, int pitch, int kb, int nt, int lane) {
    const unsigned short* p = base + (nt * 16 + (lane & 15)) * pitch + kb * 32 + ((lane & 16) ? 8 : 0);
    Vb v;
    v.u4[0] = *(const v4u*)(p);
    v.u4[1] = *(const v4u*)(p + 16);
    return v.v;
}

__device__ __forceinline__ v8f wmma_bf16(v16bf a, v16bf b, v8f c) {
    return __builtin_amdgcn_wmma_f32_16x16x32_bf16(false, a, false, b, (short)0, c, false, false);
}

// One velocity-field evaluation (primal 16-sample tile + per-sample 16-row tangent blocks).
__device__ __forceinline__ void vel_eval(Smem& S, float t, int wave, int lane,
                                         int samp, int half, int lrow, bool lo16) {
    const int kbase   = lo16 ? 0 : 8;
    const int rowbase = lo16 ? 0 : 8;
    unsigned short* thb = S.thbuf + samp * (16 * HID);
    const float* Tzs = S.Tzs + samp * 256;

    // ---------------- layer 0 (K padded 17->32) ----------------
    Vb at, ap;
#pragma unroll
    for (int i = 0; i < 8; i += 2)
        at.u32[i >> 1] = pack2bf(Tzs[lrow * 16 + kbase + i], Tzs[lrow * 16 + kbase + i + 1]);
#pragma unroll
    for (int i = 4; i < 8; ++i) at.u32[i] = 0u;
#pragma unroll
    for (int i = 0; i < 8; i += 2)
        ap.u32[i >> 1] = pack2bf(S.zs[lrow * 16 + kbase + i], S.zs[lrow * 16 + kbase + i + 1]);
    ap.u32[4] = lo16 ? pack2bf(t, 1.0f) : 0u;  // K=16 -> t, K=17 -> bias row
    ap.u32[5] = 0u; ap.u32[6] = 0u; ap.u32[7] = 0u;

    v8f ct[4];
#pragma unroll
    for (int q = 0; q < 4; ++q) {
        int nt = half * 4 + q;
        ct[q] = wmma_bf16(at.v, loadB(S.wt0, 32, 0, nt, lane), splat8(0.0f));
    }
    if (wave < 8) {
        v8f c = wmma_bf16(ap.v, loadB(S.wt0, 32, 0, wave, lane), splat8(0.0f));
        int col = wave * 16 + lrow;
#pragma unroll
        for (int r = 0; r < 8; r += 2) {
            float h0 = fast_tanh(c[r]);
            float h1 = fast_tanh(c[r + 1]);
            unsigned u = pack2bf(h0, h1);
            S.hbuf[(rowbase + r) * HID + col]     = (unsigned short)u;
            S.hbuf[(rowbase + r + 1) * HID + col] = (unsigned short)(u >> 16);
            S.dbuf[(rowbase + r) * HID + col]     = 1.0f - h0 * h0;
            S.dbuf[(rowbase + r + 1) * HID + col] = 1.0f - h1 * h1;
        }
    }
    __syncthreads();
#pragma unroll
    for (int q = 0; q < 4; ++q) {
        int col = (half * 4 + q) * 16 + lrow;
        float d = S.dbuf[samp * HID + col];
        unsigned short* tp = thb + col;
#pragma unroll
        for (int r = 0; r < 8; r += 2) {
            unsigned u = pack2bf(ct[q][r] * d, ct[q][r + 1] * d);
            tp[(rowbase + r) * HID]     = (unsigned short)u;
            tp[(rowbase + r + 1) * HID] = (unsigned short)(u >> 16);
        }
    }
    __syncthreads();

    // ---------------- layers 1..3 (K = 128) ----------------
#pragma unroll 1
    for (int l = 0; l < 3; ++l) {
        const unsigned short* wt = S.wt123[l];
        v16bf aT[4], aP[4];
#pragma unroll
        for (int kb = 0; kb < 4; ++kb) aT[kb] = loadA(thb, HID, kb, lane);
        if (wave < 8) {
#pragma unroll
            for (int kb = 0; kb < 4; ++kb) aP[kb] = loadA(S.hbuf, HID, kb, lane);
        }
        __syncthreads();
        if (wave < 8) {
            int col = wave * 16 + lrow;
            v8f c = splat8(S.bias123[l * HID + col]);
#pragma unroll
            for (int kb = 0; kb < 4; ++kb)
                c = wmma_bf16(aP[kb], loadB(wt, HID, kb, wave, lane), c);
#pragma unroll
            for (int r = 0; r < 8; r += 2) {
                float h0 = fast_tanh(c[r]);
                float h1 = fast_tanh(c[r + 1]);
                unsigned u = pack2bf(h0, h1);
                S.hbuf[(rowbase + r) * HID + col]     = (unsigned short)u;
                S.hbuf[(rowbase + r + 1) * HID + col] = (unsigned short)(u >> 16);
                S.dbuf[(rowbase + r) * HID + col]     = 1.0f - h0 * h0;
                S.dbuf[(rowbase + r + 1) * HID + col] = 1.0f - h1 * h1;
            }
        }
        v8f cq[4];
#pragma unroll
        for (int q = 0; q < 4; ++q) {
            int nt = half * 4 + q;
            v8f c = splat8(0.0f);
#pragma unroll
            for (int kb = 0; kb < 4; ++kb)
                c = wmma_bf16(aT[kb], loadB(wt, HID, kb, nt, lane), c);
            cq[q] = c;
        }
        __syncthreads();
#pragma unroll
        for (int q = 0; q < 4; ++q) {
            int col = (half * 4 + q) * 16 + lrow;
            float d = S.dbuf[samp * HID + col];
            unsigned short* tp = thb + col;
#pragma unroll
            for (int r = 0; r < 8; r += 2) {
                unsigned u = pack2bf(cq[q][r] * d, cq[q][r + 1] * d);
                tp[(rowbase + r) * HID]     = (unsigned short)u;
                tp[(rowbase + r + 1) * HID] = (unsigned short)(u >> 16);
            }
        }
        __syncthreads();
    }

    // ---------------- layer 4 (N = 16, linear) ----------------
    if (half == 0) {
        v8f c = splat8(0.0f);
#pragma unroll
        for (int kb = 0; kb < 4; ++kb)
            c = wmma_bf16(loadA(thb, HID, kb, lane), loadB(S.wt4, HID, kb, 0, lane), c);
        float* Tv = S.Tvout + samp * 256;
#pragma unroll
        for (int r = 0; r < 8; ++r) Tv[(rowbase + r) * 16 + lrow] = c[r];
    }
    if (wave == 0) {
        v8f c = splat8(S.bias4[lrow]);
#pragma unroll
        for (int kb = 0; kb < 4; ++kb)
            c = wmma_bf16(loadA(S.hbuf, HID, kb, lane), loadB(S.wt4, HID, kb, 0, lane), c);
#pragma unroll
        for (int r = 0; r < 8; ++r) S.vout[(rowbase + r) * 16 + lrow] = c[r];
    }
}

__global__ __launch_bounds__(THREADS)
void fm_metric_kernel(const float* __restrict__ x,  const float* __restrict__ xst,
                      const float* __restrict__ W0, const float* __restrict__ b0,
                      const float* __restrict__ W1, const float* __restrict__ b1,
                      const float* __restrict__ W2, const float* __restrict__ b2,
                      const float* __restrict__ W3, const float* __restrict__ b3,
                      const float* __restrict__ W4, const float* __restrict__ b4,
                      float* __restrict__ out) {
    extern __shared__ char smem_raw[];
    Smem& S = *reinterpret_cast<Smem*>(smem_raw);

    const int tid  = threadIdx.x;
    const int wave = tid >> 5;
    const int lane = tid & 31;
    const int samp = wave >> 1;
    const int half = wave & 1;
    const int lrow = lane & 15;
    const bool lo16 = lane < 16;
    const int blk8 = blockIdx.x * SAMP_PER_WG;

    // ---- stage weights into LDS: transposed, bf16 ----
    for (int i = tid; i < HID * 32; i += THREADS) {
        int j = i >> 5, k = i & 31;
        float v = 0.0f;
        if (k < 17)       v = W0[k * HID + j];  // includes t-row (k==16)
        else if (k == 17) v = b0[j];            // bias folded via ones column
        S.wt0[j * 32 + k] = f2bf(v);
    }
    const float* Wm[3] = {W1, W2, W3};
    const float* bm[3] = {b1, b2, b3};
#pragma unroll 1
    for (int l = 0; l < 3; ++l) {
        for (int i = tid; i < HID * HID; i += THREADS) {
            int j = i >> 7, k = i & 127;
            S.wt123[l][j * HID + k] = f2bf(Wm[l][k * HID + j]);
        }
        for (int i = tid; i < HID; i += THREADS) S.bias123[l * HID + i] = bm[l][i];
    }
    for (int i = tid; i < DIM * HID; i += THREADS) {
        int j = i >> 7, k = i & 127;
        S.wt4[j * HID + k] = f2bf(W4[k * DIM + j]);
    }
    if (tid < DIM) S.bias4[tid] = b4[tid];

    // ---- init RK4 state: rows 0-7 = x, rows 8-15 = x_star; tangents = I ----
    for (int i = tid; i < 256; i += THREADS) {
        int r = i >> 4, c = i & 15;
        float v = (r < 8) ? x[(blk8 + r) * DIM + c] : xst[(blk8 + r - 8) * DIM + c];
        S.z[i] = v;
        S.zs[i] = v;
        S.ka[i] = 0.0f;
    }
    for (int i = tid; i < 2048; i += THREADS) {
        int r = (i >> 4) & 15, c = i & 15;
        float v = (r == c) ? 1.0f : 0.0f;
        S.Tz[i] = v; S.Tzs[i] = v; S.Tka[i] = 0.0f;
    }

    const float dt = 1.0f / STEPS;
#pragma unroll 1
    for (int step = 0; step < STEPS; ++step) {
        float tb = step * dt;
#pragma unroll 1
        for (int st = 0; st < 4; ++st) {
            const float toff = (st == 0) ? 0.0f : ((st == 3) ? 1.0f : 0.5f);
            const float wac  = (st == 0 || st == 3) ? 1.0f : 2.0f;
            const float nxt  = (st < 2) ? 0.5f : 1.0f;  // stage-input coefficient for st+1
            __syncthreads();
            vel_eval(S, tb + toff * dt, wave, lane, samp, half, lrow, lo16);
            __syncthreads();
            for (int i = tid; i < 256; i += THREADS) {
                float v = S.vout[i];
                S.ka[i] += wac * v;
                if (st < 3) S.zs[i] = S.z[i] + nxt * dt * v;
                else        { S.z[i] += (dt / 6.0f) * S.ka[i]; S.ka[i] = 0.0f; }
            }
            for (int i = tid; i < 2048; i += THREADS) {
                float v = S.Tvout[i];
                S.Tka[i] += wac * v;
                if (st < 3) S.Tzs[i] = S.Tz[i] + nxt * dt * v;
                else        { S.Tz[i] += (dt / 6.0f) * S.Tka[i]; S.Tka[i] = 0.0f; }
            }
        }
        // refresh stage inputs for next step
        __syncthreads();
        for (int i = tid; i < 256; i += THREADS)  S.zs[i]  = S.z[i];
        for (int i = tid; i < 2048; i += THREADS) S.Tzs[i] = S.Tz[i];
    }
    __syncthreads();

    // ---- per-sample: g, rhs = Th*g, G = Th*Th^T + eps, SPD solve, out = -u ----
    if (half == 0 && lane < 16) {
        const int j = lane;
        const float* Ts = S.Tz + samp * 256;  // Th = J^T, row j = d(out)/d(x_j)
        float th[16];
#pragma unroll
        for (int i = 0; i < 16; ++i) th[i] = Ts[j * 16 + i];
        float g[16], nn = 0.0f;
#pragma unroll
        for (int i = 0; i < 16; ++i) {
            float d = S.z[samp * 16 + i] - S.z[(samp + 8) * 16 + i];
            g[i] = d; nn += d * d;
        }
        float inv = 1.0f / (sqrtf(nn) + 1e-8f);
        float rhs = 0.0f;
#pragma unroll
        for (int i = 0; i < 16; ++i) rhs += th[i] * (g[i] * inv);
        float G[16];
#pragma unroll
        for (int k = 0; k < 16; ++k) {
            float acc = 0.0f;
#pragma unroll
            for (int i = 0; i < 16; ++i) acc += th[i] * __shfl(th[i], k, 32);
            G[k] = acc;
        }
        G[j] += 1e-6f;
        // forward elimination (SPD, no pivoting)
#pragma unroll
        for (int k = 0; k < 16; ++k) {
            float pk = __shfl(G[k], k, 32);
            float pr = __shfl(rhs, k, 32);
            float f = (j > k) ? (G[k] / pk) : 0.0f;
#pragma unroll
            for (int i = 0; i < 16; ++i) G[i] -= f * __shfl(G[i], k, 32);
            rhs -= f * pr;
        }
        // back substitution
        float sol = 0.0f;
#pragma unroll
        for (int k = 15; k >= 0; --k) {
            float xk = __shfl(rhs, k, 32) / __shfl(G[k], k, 32);
            if (j == k) sol = xk;
            if (j < k) rhs -= G[k] * xk;
        }
        out[(blk8 + samp) * DIM + j] = -sol;
    }
}

extern "C" void kernel_launch(void* const* d_in, const int* in_sizes, int n_in,
                              void* d_out, int out_size, void* d_ws, size_t ws_size,
                              hipStream_t stream) {
    (void)in_sizes; (void)n_in; (void)d_ws; (void)ws_size; (void)out_size;
    const float* x   = (const float*)d_in[0];
    const float* xst = (const float*)d_in[1];
    const float* W0  = (const float*)d_in[2];
    const float* b0  = (const float*)d_in[3];
    const float* W1  = (const float*)d_in[4];
    const float* b1  = (const float*)d_in[5];
    const float* W2  = (const float*)d_in[6];
    const float* b2  = (const float*)d_in[7];
    const float* W3  = (const float*)d_in[8];
    const float* b3  = (const float*)d_in[9];
    const float* W4  = (const float*)d_in[10];
    const float* b4  = (const float*)d_in[11];
    float* out = (float*)d_out;

    const size_t smem = sizeof(Smem);
    hipFuncSetAttribute((const void*)fm_metric_kernel,
                        hipFuncAttributeMaxDynamicSharedMemorySize, (int)smem);
    fm_metric_kernel<<<NBLK, THREADS, smem, stream>>>(
        x, xst, W0, b0, W1, b1, W2, b2, W3, b3, W4, b4, out);
}
